// BinaryClassifier_79551384256625
// MI455X (gfx1250) — matrix operand order repaired and checked by reading, not compiled, not run on hardware
//
#include <hip/hip_runtime.h>
#include <math.h>

// ---------------------------------------------------------------------------
// EGAT (3 layers) + node MLP + T*T pairwise MLP classifier for gfx1250.
// All GEMMs: v_wmma_f32_16x16x32_bf16, bf16 operands pre-packed (activations
// converted once; weights transposed to N x K bf16 so B-fragments are
// contiguous b128 loads).  Edge softmax via float atomic max/add; aggregation
// via global_atomic_add_f32 scatter.
// ---------------------------------------------------------------------------

#define N_NODES 20000
#define N_EDGES 320000
#define T_TRIP  1536
#define HD      128   // H*D = 4*32
#define NODE_IN 64

typedef __bf16 bf16_t;
typedef bf16_t v16bf __attribute__((ext_vector_type(16)));
typedef bf16_t v8bf  __attribute__((ext_vector_type(8)));
typedef float  v8f   __attribute__((ext_vector_type(8)));

__device__ __forceinline__ bf16_t f32_to_bf16(float f) {
  unsigned u = __builtin_bit_cast(unsigned, f);
  unsigned r = (u + 0x7FFFu + ((u >> 16) & 1u)) >> 16;   // round-to-nearest-even
  return __builtin_bit_cast(bf16_t, (unsigned short)r);
}

__device__ __forceinline__ void atomic_max_f32(float* addr, float val) {
  if (val >= 0.0f)
    atomicMax((int*)addr, __float_as_int(val));
  else
    atomicMin((unsigned int*)addr, (unsigned int)__float_as_int(val));
}

__device__ __forceinline__ v16bf cat8(v8bf lo, v8bf hi) {
  return __builtin_shufflevector(lo, hi, 0, 1, 2, 3, 4, 5, 6, 7,
                                 8, 9, 10, 11, 12, 13, 14, 15);
}

// ---------------------------------------------------------------------------
// WMMA GEMM: out[M x Ncols](f32) = act(A[M x K](bf16) * Wt[Ncols x K](bf16)^T
//                                      + bias)
// M mult of 16 (grid.x = M/16), K mult of 32, Ncols mult of 16.
// Block = 256 = 8 waves; wave w covers column tiles w, w+8, ...
// act: 0 = none, 1 = relu
// ---------------------------------------------------------------------------
__global__ __launch_bounds__(256)
void gemm_bf16(const bf16_t* __restrict__ A, const bf16_t* __restrict__ Wt,
               const float* __restrict__ bias, float* __restrict__ out,
               int K, int Ncols, int act) {
  const int lane  = threadIdx.x & 31;
  const int wave  = threadIdx.x >> 5;
  const int lo    = lane & 15;
  const int hi    = lane >> 4;
  const int row0  = blockIdx.x << 4;
  const int kbase = hi << 3;   // A-frag K base within each 16-half
  const int khalf = hi << 4;   // unused; B-frag loads use kbase, same K order as A
  const bf16_t* arow = A + (size_t)(row0 + lo) * K;

  for (int n0 = wave << 4; n0 < Ncols; n0 += 128) {
    const int ncol = n0 + lo;
    const bf16_t* wrow = Wt + (size_t)ncol * K;
    float bv = bias ? bias[ncol] : 0.0f;
    v8f c;
#pragma unroll
    for (int r = 0; r < 8; ++r) c[r] = bv;

    for (int k0 = 0; k0 < K; k0 += 32) {
      v8bf a_lo = *(const v8bf*)(arow + k0 + kbase);        // K = kbase..+7
      v8bf a_hi = *(const v8bf*)(arow + k0 + 16 + kbase);   // K = 16+kbase..
      v8bf b_lo = *(const v8bf*)(wrow + k0 + kbase);        // K = kbase..+7
      v8bf b_hi = *(const v8bf*)(wrow + k0 + 16 + kbase);   // K = 16+kbase..
      c = __builtin_amdgcn_wmma_f32_16x16x32_bf16(false, cat8(a_lo, a_hi),
                                                  false, cat8(b_lo, b_hi),
                                                  (short)0, c, false, false);
    }
#pragma unroll
    for (int r = 0; r < 8; ++r) {
      float x = c[r];
      if (act == 1) x = x > 0.0f ? x : 0.0f;
      out[(size_t)(row0 + r + (hi << 3)) * Ncols + ncol] = x;
    }
  }
}

// ---------------------------------------------------------------------------
__global__ void fill_f32(float* __restrict__ p, float v, long long n) {
  long long i = (long long)blockIdx.x * blockDim.x + threadIdx.x;
  if (i < n) p[i] = v;
}

// f32 -> bf16 elementwise
__global__ void conv_bf16(const float* __restrict__ in,
                          bf16_t* __restrict__ outp, long long n) {
  long long g = (long long)blockIdx.x * blockDim.x + threadIdx.x;
  if (g < n) outp[g] = f32_to_bf16(in[g]);
}

// W[K x Ncols] f32  ->  Wt[Ncols x K] bf16 (transposed repack)
__global__ void repack_w(const float* __restrict__ W, bf16_t* __restrict__ Wt,
                         int K, int Ncols) {
  int g = blockIdx.x * blockDim.x + threadIdx.x;
  if (g >= K * Ncols) return;
  int n = g / K, k = g % K;
  Wt[g] = f32_to_bf16(W[(size_t)k * Ncols + n]);
}

// ---------------------------------------------------------------------------
// Edge update: f = LReLU(ni[src] + nj[dst] + fij + b); e[h] = <f_h, attn_h>
// One thread per (edge, head).  f stored as bf16 (next layer's GEMM input).
// ---------------------------------------------------------------------------
__global__ __launch_bounds__(256)
void egat_edge(const float* __restrict__ ni, const float* __restrict__ nj,
               const float* __restrict__ fij, const float* __restrict__ b_edge,
               const float* __restrict__ attn,
               const int* __restrict__ src, const int* __restrict__ dst,
               bf16_t* __restrict__ f_out, float* __restrict__ e_out,
               float* __restrict__ mbuf) {
  int g = blockIdx.x * blockDim.x + threadIdx.x;
  if (g >= N_EDGES * 4) return;
  int e = g >> 2, h = g & 3;
  int s = src[e], d = dst[e];
  const float* pi = ni  + (size_t)s * HD + h * 32;
  const float* pj = nj  + (size_t)d * HD + h * 32;
  const float* pf = fij + (size_t)e * HD + h * 32;
  const float* pb = b_edge + h * 32;
  const float* pa = attn   + h * 32;
  bf16_t* po = f_out + (size_t)e * HD + h * 32;
  float acc = 0.0f;
#pragma unroll 8
  for (int t = 0; t < 32; ++t) {
    float f = pi[t] + pj[t] + pf[t] + pb[t];
    f = f > 0.0f ? f : 0.01f * f;          // leaky_relu (slope 0.01)
    po[t] = f32_to_bf16(f);
    acc += f * pa[t];
  }
  e_out[g] = acc;
  atomic_max_f32(&mbuf[d * 4 + h], acc);
}

__global__ __launch_bounds__(256)
void egat_exp(const float* __restrict__ e_out, const float* __restrict__ mbuf,
              const int* __restrict__ dst,
              float* __restrict__ a_out, float* __restrict__ sbuf) {
  int g = blockIdx.x * blockDim.x + threadIdx.x;
  if (g >= N_EDGES * 4) return;
  int e = g >> 2, h = g & 3;
  int d = dst[e];
  float a = __expf(e_out[g] - mbuf[d * 4 + h]);
  a_out[g] = a;
  atomicAdd(&sbuf[d * 4 + h], a);
}

__global__ __launch_bounds__(256)
void egat_scatter(const float* __restrict__ hs, const float* __restrict__ a_out,
                  const float* __restrict__ sbuf,
                  const int* __restrict__ src, const int* __restrict__ dst,
                  float* __restrict__ h_out) {
  int g = blockIdx.x * blockDim.x + threadIdx.x;
  if (g >= N_EDGES * 4) return;
  int e = g >> 2, h = g & 3;
  int s = src[e], d = dst[e];
  float a = a_out[g] / sbuf[d * 4 + h];
  const float* ph = hs + (size_t)s * HD + h * 32;
  float* po = h_out + (size_t)d * HD + h * 32;
#pragma unroll 8
  for (int t = 0; t < 32; ++t) atomicAdd(&po[t], ph[t] * a);
}

// ---------------------------------------------------------------------------
// cat[t,0:32] = mean over 4 heads of nfeat3[t];  cat[t,32:96] = inputs[t]
// Written directly in bf16 (consumed only by GEMM).
// ---------------------------------------------------------------------------
__global__ void mean_cat(const float* __restrict__ nf,
                         const float* __restrict__ inputs,
                         bf16_t* __restrict__ cat) {
  int g = blockIdx.x * blockDim.x + threadIdx.x;
  if (g >= T_TRIP * 96) return;
  int row = g / 96, col = g % 96;
  float v;
  if (col < 32) {
    const float* p = nf + (size_t)row * HD;
    v = 0.25f * (p[col] + p[32 + col] + p[64 + col] + p[96 + col]);
  } else {
    v = inputs[(size_t)row * NODE_IN + (col - 32)];
  }
  cat[g] = f32_to_bf16(v);
}

// ---------------------------------------------------------------------------
// Pairwise MLP: for pair k = i*T + j:
//   x = [h[i](16) | h[j](16)]; relu(x W1+b1); relu(. W2+b2); relu(. W3+b3);
//   out = sigmoid(. w4 + b4)
// One block per i; 8 waves, each wave handles 12 j-tiles of 16.
// 5 bf16 WMMAs per tile; inter-layer transpose staged via per-wave LDS.
// ---------------------------------------------------------------------------
__global__ __launch_bounds__(256)
void pairwise(const bf16_t* __restrict__ h2,
              const float* __restrict__ W1, const float* __restrict__ B1,
              const float* __restrict__ W2, const float* __restrict__ B2,
              const float* __restrict__ W3, const float* __restrict__ B3,
              const float* __restrict__ W4, const float* __restrict__ B4,
              float* __restrict__ out) {
  __shared__ bf16_t stage[8][16 * 32];
  const int lane  = threadIdx.x & 31;
  const int wave  = threadIdx.x >> 5;
  const int i     = blockIdx.x;
  const int lo    = lane & 15;
  const int hi    = lane >> 4;
  const int kbase = hi << 3;
  const int khalf = hi << 4;
  bf16_t* myst = stage[wave];

  // Hoisted B-fragments (constant weights)
  v16bf w1a, w1b, w2a, w2b, w3f;
#pragma unroll
  for (int t = 0; t < 16; ++t) {
    w1a[t] = f32_to_bf16(W1[(kbase + (t & 7) + ((t >> 3) << 4)) * 32 + lo]);
    w1b[t] = f32_to_bf16(W1[(kbase + (t & 7) + ((t >> 3) << 4)) * 32 + 16 + lo]);
    w2a[t] = f32_to_bf16(W2[(kbase + (t & 7) + ((t >> 3) << 4)) * 32 + lo]);
    w2b[t] = f32_to_bf16(W2[(kbase + (t & 7) + ((t >> 3) << 4)) * 32 + 16 + lo]);
    w3f[t] = f32_to_bf16(W3[(kbase + (t & 7) + ((t >> 3) << 4)) * 16 + lo]);
  }
  const float b1a = B1[lo], b1b = B1[16 + lo];
  const float b2a = B2[lo], b2b = B2[16 + lo];
  const float b3v = B3[lo];
  const float w4v = W4[lo];
  const float b4v = B4[0];

  // h[i] half of the A fragment (K = 0..15), constant across tiles
  v8bf ai = *(const v8bf*)(h2 + i * 16 + kbase);

  for (int jt = wave; jt < T_TRIP / 16; jt += 8) {
    const int j0 = jt << 4;
    // ---- layer 1: A = [h[i] | h[j0+m]] (16x32), W1 (32x32) ----
    v8bf aj = *(const v8bf*)(h2 + (size_t)(j0 + lo) * 16 + kbase);
    v16bf a1 = cat8(ai, aj);

    v8f c1a, c1b;
#pragma unroll
    for (int r = 0; r < 8; ++r) { c1a[r] = b1a; c1b[r] = b1b; }
    c1a = __builtin_amdgcn_wmma_f32_16x16x32_bf16(false, a1, false, w1a, (short)0, c1a, false, false);
    c1b = __builtin_amdgcn_wmma_f32_16x16x32_bf16(false, a1, false, w1b, (short)0, c1b, false, false);

    // relu -> bf16 -> LDS (row-major 16x32), then reload as A fragment
#pragma unroll
    for (int r = 0; r < 8; ++r) {
      int m = r + (hi << 3);
      float x = c1a[r]; x = x > 0.0f ? x : 0.0f;
      myst[m * 32 + lo] = f32_to_bf16(x);
      x = c1b[r]; x = x > 0.0f ? x : 0.0f;
      myst[m * 32 + 16 + lo] = f32_to_bf16(x);
    }
    asm volatile("s_wait_dscnt 0x0" ::: "memory");
    v16bf a2 = cat8(*(const v8bf*)(myst + lo * 32 + kbase),
                    *(const v8bf*)(myst + lo * 32 + 16 + kbase));
    asm volatile("s_wait_dscnt 0x0" ::: "memory");

    // ---- layer 2 ----
    v8f c2a, c2b;
#pragma unroll
    for (int r = 0; r < 8; ++r) { c2a[r] = b2a; c2b[r] = b2b; }
    c2a = __builtin_amdgcn_wmma_f32_16x16x32_bf16(false, a2, false, w2a, (short)0, c2a, false, false);
    c2b = __builtin_amdgcn_wmma_f32_16x16x32_bf16(false, a2, false, w2b, (short)0, c2b, false, false);

#pragma unroll
    for (int r = 0; r < 8; ++r) {
      int m = r + (hi << 3);
      float x = c2a[r]; x = x > 0.0f ? x : 0.0f;
      myst[m * 32 + lo] = f32_to_bf16(x);
      x = c2b[r]; x = x > 0.0f ? x : 0.0f;
      myst[m * 32 + 16 + lo] = f32_to_bf16(x);
    }
    asm volatile("s_wait_dscnt 0x0" ::: "memory");
    v16bf a3 = cat8(*(const v8bf*)(myst + lo * 32 + kbase),
                    *(const v8bf*)(myst + lo * 32 + 16 + kbase));
    asm volatile("s_wait_dscnt 0x0" ::: "memory");

    // ---- layer 3 (32 -> 16) ----
    v8f c3;
#pragma unroll
    for (int r = 0; r < 8; ++r) c3[r] = b3v;
    c3 = __builtin_amdgcn_wmma_f32_16x16x32_bf16(false, a3, false, w3f, (short)0, c3, false, false);

    // ---- layer 4 (16 -> 1) + sigmoid: cross-lane dot within 16-lane halves
    float* op = out + (size_t)i * T_TRIP + j0;
#pragma unroll
    for (int r = 0; r < 8; ++r) {
      float y = c3[r]; y = y > 0.0f ? y : 0.0f;   // relu(l3)
      float p = y * w4v;
      p += __shfl_xor(p, 1);
      p += __shfl_xor(p, 2);
      p += __shfl_xor(p, 4);
      p += __shfl_xor(p, 8);
      if (lo == 0) {
        int m = r + (hi << 3);
        op[m] = 1.0f / (1.0f + __expf(-(p + b4v)));
      }
    }
  }
}

// ---------------------------------------------------------------------------
// d_in flattening assumption: setup_inputs() dict insertion order, recursive:
//  0 inputs, 1 edge_feats,
//  2..7   egat1 {w_ni,w_nj,w_fij,attn,b_edge,w_node}
//  8..13  egat2, 14..19 egat3,
//  20/21 ml1.w/b, 22/23 ml2, 24/25 ml3,
//  26/27 l1, 28/29 l2, 30/31 l3, 32/33 l4,
//  34 src, 35 dst, 36 mask, 37 T
// ---------------------------------------------------------------------------
extern "C" void kernel_launch(void* const* d_in, const int* in_sizes, int n_in,
                              void* d_out, int out_size, void* d_ws, size_t ws_size,
                              hipStream_t stream) {
  const float* inputs     = (const float*)d_in[0];
  const float* edge_feats = (const float*)d_in[1];
  const int*   src        = (const int*)d_in[34];
  const int*   dst        = (const int*)d_in[35];

  const float* eg_wni[3]   = {(const float*)d_in[2],  (const float*)d_in[8],  (const float*)d_in[14]};
  const float* eg_wnj[3]   = {(const float*)d_in[3],  (const float*)d_in[9],  (const float*)d_in[15]};
  const float* eg_wfij[3]  = {(const float*)d_in[4],  (const float*)d_in[10], (const float*)d_in[16]};
  const float* eg_attn[3]  = {(const float*)d_in[5],  (const float*)d_in[11], (const float*)d_in[17]};
  const float* eg_b[3]     = {(const float*)d_in[6],  (const float*)d_in[12], (const float*)d_in[18]};
  const float* eg_wnode[3] = {(const float*)d_in[7],  (const float*)d_in[13], (const float*)d_in[19]};

  // ---- workspace carve-out -------------------------------------------------
  char* wsp = (char*)d_ws;
  size_t off = 0;
  auto take = [&](size_t bytes) -> void* {
    void* p = wsp + off;
    off += (bytes + 255) & ~(size_t)255;
    return p;
  };
  // f32 buffers
  float* ni   = (float*)take((size_t)N_NODES * HD * 4);
  float* nj   = (float*)take((size_t)N_NODES * HD * 4);
  float* hs   = (float*)take((size_t)N_NODES * HD * 4);
  float* nA   = (float*)take((size_t)N_NODES * HD * 4);
  float* nB   = (float*)take((size_t)N_NODES * HD * 4);
  float* fij  = (float*)take((size_t)N_EDGES * HD * 4);
  float* ebuf = (float*)take((size_t)N_EDGES * 4 * 4);
  float* abuf = (float*)take((size_t)N_EDGES * 4 * 4);
  float* mbuf = (float*)take((size_t)N_NODES * 4 * 4);
  float* sbuf = (float*)take((size_t)N_NODES * 4 * 4);
  float* t64  = (float*)take((size_t)T_TRIP * 64 * 4);
  float* t32  = (float*)take((size_t)T_TRIP * 32 * 4);
  float* t16  = (float*)take((size_t)T_TRIP * 16 * 4);
  // bf16 buffers
  bf16_t* inb  = (bf16_t*)take((size_t)N_NODES * NODE_IN * 2);
  bf16_t* ef0b = (bf16_t*)take((size_t)N_EDGES * 32 * 2);
  bf16_t* nAb  = (bf16_t*)take((size_t)N_NODES * HD * 2);
  bf16_t* nBb  = (bf16_t*)take((size_t)N_NODES * HD * 2);
  bf16_t* fAb  = (bf16_t*)take((size_t)N_EDGES * HD * 2);
  bf16_t* fBb  = (bf16_t*)take((size_t)N_EDGES * HD * 2);
  bf16_t* catb = (bf16_t*)take((size_t)T_TRIP * 96 * 2);
  bf16_t* t64b = (bf16_t*)take((size_t)T_TRIP * 64 * 2);
  bf16_t* t32b = (bf16_t*)take((size_t)T_TRIP * 32 * 2);
  bf16_t* h2   = (bf16_t*)take((size_t)T_TRIP * 16 * 2);
  // transposed bf16 weights
  bf16_t* wni_t[3];
  bf16_t* wnj_t[3];
  bf16_t* wnode_t[3];
  bf16_t* wfij_t[3];
  const int in_n[3] = {NODE_IN, HD, HD};
  const int in_e[3] = {32, HD, HD};
  for (int l = 0; l < 3; ++l) {
    wni_t[l]   = (bf16_t*)take((size_t)in_n[l] * HD * 2);
    wnj_t[l]   = (bf16_t*)take((size_t)in_n[l] * HD * 2);
    wnode_t[l] = (bf16_t*)take((size_t)in_n[l] * HD * 2);
    wfij_t[l]  = (bf16_t*)take((size_t)in_e[l] * HD * 2);
  }
  bf16_t* ml1_t = (bf16_t*)take((size_t)96 * 64 * 2);
  bf16_t* ml2_t = (bf16_t*)take((size_t)64 * 32 * 2);
  bf16_t* ml3_t = (bf16_t*)take((size_t)32 * 16 * 2);

  // ---- weight repacks + activation conversions ----------------------------
  for (int l = 0; l < 3; ++l) {
    int kn = in_n[l], ke = in_e[l];
    repack_w<<<(kn * HD + 255) / 256, 256, 0, stream>>>(eg_wni[l],   wni_t[l],   kn, HD);
    repack_w<<<(kn * HD + 255) / 256, 256, 0, stream>>>(eg_wnj[l],   wnj_t[l],   kn, HD);
    repack_w<<<(kn * HD + 255) / 256, 256, 0, stream>>>(eg_wnode[l], wnode_t[l], kn, HD);
    repack_w<<<(ke * HD + 255) / 256, 256, 0, stream>>>(eg_wfij[l],  wfij_t[l],  ke, HD);
  }
  repack_w<<<(96 * 64 + 255) / 256, 256, 0, stream>>>((const float*)d_in[20], ml1_t, 96, 64);
  repack_w<<<(64 * 32 + 255) / 256, 256, 0, stream>>>((const float*)d_in[22], ml2_t, 64, 32);
  repack_w<<<(32 * 16 + 255) / 256, 256, 0, stream>>>((const float*)d_in[24], ml3_t, 32, 16);
  conv_bf16<<<(unsigned)(((long long)N_NODES * NODE_IN + 255) / 256), 256, 0, stream>>>(
      inputs, inb, (long long)N_NODES * NODE_IN);
  conv_bf16<<<(unsigned)(((long long)N_EDGES * 32 + 255) / 256), 256, 0, stream>>>(
      edge_feats, ef0b, (long long)N_EDGES * 32);

  // ---- 3 EGAT layers -------------------------------------------------------
  const bf16_t* nfeat_b[3] = {inb, nAb, nBb};
  const bf16_t* efeat_b[3] = {ef0b, fAb, fBb};
  bf16_t*       f_out_b[3] = {fAb, fBb, fAb};
  float*        h_out[3]   = {nA, nB, nA};
  bf16_t*       h_out_b[3] = {nAb, nBb, nullptr};

  const int EH = N_EDGES * 4;
  for (int l = 0; l < 3; ++l) {
    gemm_bf16<<<N_NODES / 16, 256, 0, stream>>>(nfeat_b[l], wni_t[l],   nullptr, ni,  in_n[l], HD, 0);
    gemm_bf16<<<N_NODES / 16, 256, 0, stream>>>(nfeat_b[l], wnj_t[l],   nullptr, nj,  in_n[l], HD, 0);
    gemm_bf16<<<N_NODES / 16, 256, 0, stream>>>(nfeat_b[l], wnode_t[l], nullptr, hs,  in_n[l], HD, 0);
    gemm_bf16<<<N_EDGES / 16, 256, 0, stream>>>(efeat_b[l], wfij_t[l],  nullptr, fij, in_e[l], HD, 0);

    fill_f32<<<(N_NODES * 4 + 255) / 256, 256, 0, stream>>>(mbuf, -INFINITY, (long long)N_NODES * 4);
    fill_f32<<<(N_NODES * 4 + 255) / 256, 256, 0, stream>>>(sbuf, 0.0f, (long long)N_NODES * 4);
    fill_f32<<<(unsigned)(((long long)N_NODES * HD + 255) / 256), 256, 0, stream>>>(
        h_out[l], 0.0f, (long long)N_NODES * HD);

    egat_edge<<<(EH + 255) / 256, 256, 0, stream>>>(ni, nj, fij, eg_b[l], eg_attn[l],
                                                    src, dst, f_out_b[l], ebuf, mbuf);
    egat_exp<<<(EH + 255) / 256, 256, 0, stream>>>(ebuf, mbuf, dst, abuf, sbuf);
    egat_scatter<<<(EH + 255) / 256, 256, 0, stream>>>(hs, abuf, sbuf, src, dst, h_out[l]);
    if (h_out_b[l])
      conv_bf16<<<(unsigned)(((long long)N_NODES * HD + 255) / 256), 256, 0, stream>>>(
          h_out[l], h_out_b[l], (long long)N_NODES * HD);
  }

  // ---- node MLP on first T nodes (mask = arange(N) < T) --------------------
  mean_cat<<<(T_TRIP * 96 + 255) / 256, 256, 0, stream>>>(h_out[2], inputs, catb);
  gemm_bf16<<<T_TRIP / 16, 256, 0, stream>>>(catb, ml1_t, (const float*)d_in[21], t64, 96, 64, 1);
  conv_bf16<<<(T_TRIP * 64 + 255) / 256, 256, 0, stream>>>(t64, t64b, (long long)T_TRIP * 64);
  gemm_bf16<<<T_TRIP / 16, 256, 0, stream>>>(t64b, ml2_t, (const float*)d_in[23], t32, 64, 32, 1);
  conv_bf16<<<(T_TRIP * 32 + 255) / 256, 256, 0, stream>>>(t32, t32b, (long long)T_TRIP * 32);
  gemm_bf16<<<T_TRIP / 16, 256, 0, stream>>>(t32b, ml3_t, (const float*)d_in[25], t16, 32, 16, 0);
  conv_bf16<<<(T_TRIP * 16 + 255) / 256, 256, 0, stream>>>(t16, h2, (long long)T_TRIP * 16);

  // ---- pairwise classifier over T*T pairs ----------------------------------
  pairwise<<<T_TRIP, 256, 0, stream>>>(h2,
                                       (const float*)d_in[26], (const float*)d_in[27],
                                       (const float*)d_in[28], (const float*)d_in[29],
                                       (const float*)d_in[30], (const float*)d_in[31],
                                       (const float*)d_in[32], (const float*)d_in[33],
                                       (float*)d_out);
}
